// MultiHeadAttention_74947179316274
// MI455X (gfx1250) — compile-verified, matrix operand order checked by reading, not run on hardware
//
#include <hip/hip_runtime.h>
#include <hip/hip_bf16.h>
#include <cstdint>
#include <cstddef>

// MHA: B=8, S=1024, H=1024, NH=16, HD=64
#define B_   8
#define S_   1024
#define H_   1024
#define NH_  16
#define HD_  64
#define NEGV (-1e32f)

typedef __attribute__((ext_vector_type(16))) __bf16 v16bf;
typedef __attribute__((ext_vector_type(8)))  __bf16 v8bf;
typedef __attribute__((ext_vector_type(8)))  float  v8f;
typedef __attribute__((ext_vector_type(4)))  float  v4f;

static __device__ inline v8f wmma_bf16(v16bf a, v16bf b, v8f c) {
  // D = A(16x32 bf16) * B(32x16 bf16) + C(16x16 f32)
  return __builtin_amdgcn_wmma_f32_16x16x32_bf16(false, a, false, b, (short)0, c,
                                                 false, false);
}

static __device__ inline v8f v8f_zero() {
  v8f z;
#pragma unroll
  for (int i = 0; i < 8; ++i) z[i] = 0.0f;
  return z;
}

// A/B fragment gather for 16-bit WMMA (wave32 layout):
// lane L (hf = L/16): element i -> k = k0 + (i/8)*16 + hf*8 + (i%8)
// => two contiguous 8-element (16-byte) chunks per lane.
static __device__ inline v16bf frag_bf16(const __bf16* __restrict__ row, int k0, int hf) {
  v8bf c0 = *(const v8bf*)(row + k0 + hf * 8);
  v8bf c1 = *(const v8bf*)(row + k0 + 16 + hf * 8);
  return __builtin_shufflevector(c0, c1, 0, 1, 2, 3, 4, 5, 6, 7,
                                 8, 9, 10, 11, 12, 13, 14, 15);
}

static __device__ inline v16bf frag_f32(const float* __restrict__ row, int k0, int hf) {
  const v4f* p0 = (const v4f*)(row + k0 + hf * 8);
  const v4f* p1 = (const v4f*)(row + k0 + 16 + hf * 8);
  v4f x0 = p0[0], x1 = p0[1], y0 = p1[0], y1 = p1[1];
  v16bf r;
#pragma unroll
  for (int i = 0; i < 4; ++i) {
    r[i]      = (__bf16)x0[i];
    r[4 + i]  = (__bf16)x1[i];
    r[8 + i]  = (__bf16)y0[i];
    r[12 + i] = (__bf16)y1[i];
  }
  return r;
}

// ---------------------------------------------------------------------------
// Kernel 0: one-shot fp32 -> bf16 conversion (used for the 4 weight matrices,
// so GEMM inner loops consume pure bf16 B-fragments with zero VALU cvt work).
// ---------------------------------------------------------------------------
__global__ __launch_bounds__(256) void cvt_f32_to_bf16_kernel(
    const float* __restrict__ src, __bf16* __restrict__ dst) {
  const int i = (blockIdx.x * 256 + threadIdx.x) * 8;
  v4f a = *(const v4f*)(src + i);
  v4f b = *(const v4f*)(src + i + 4);
  v8bf p;
#pragma unroll
  for (int j = 0; j < 4; ++j) {
    p[j]     = (__bf16)a[j];
    p[4 + j] = (__bf16)b[j];
  }
  *(v8bf*)(dst + i) = p;
}

// ---------------------------------------------------------------------------
// Kernel 1: fused QKV projection.  y[b,s,o] = sum_h x[b,s,h] * W[o,h] + bias[o]
// grid = (BS/128, H/128, 3); block = 256 (8 waves, each a 16x128 strip).
// W is pre-converted bf16. Q,K written bf16 as [b, h, s, d];
// V written bf16 TRANSPOSED as [b, h, d, s].
// ---------------------------------------------------------------------------
__global__ __launch_bounds__(256) void qkv_proj_kernel(
    const float* __restrict__ q_in, const float* __restrict__ k_in,
    const float* __restrict__ v_in,
    const __bf16* __restrict__ Wq, const float* __restrict__ bq,
    const __bf16* __restrict__ Wk, const float* __restrict__ bk,
    const __bf16* __restrict__ Wv, const float* __restrict__ bv,
    __bf16* __restrict__ qw, __bf16* __restrict__ kw, __bf16* __restrict__ vtw) {
  const int tid = threadIdx.x, wave = tid >> 5, lane = tid & 31;
  const int hf = lane >> 4, l16 = lane & 15;
  const int z = blockIdx.z;

  const float*  X    = (z == 0) ? q_in : ((z == 1) ? k_in : v_in);
  const __bf16* W    = (z == 0) ? Wq   : ((z == 1) ? Wk   : Wv);
  const float*  bias = (z == 0) ? bq   : ((z == 1) ? bk   : bv);

  const int m0 = blockIdx.x * 128 + wave * 16;  // global row in [0, B*S)
  const int n0 = blockIdx.y * 128;              // output column base

  const float* arow = X + (size_t)(m0 + l16) * H_;

  const __bf16* wrow[8];
  float bcol[8];
#pragma unroll
  for (int j = 0; j < 8; ++j) {
    int n = n0 + j * 16 + l16;
    wrow[j] = W + (size_t)n * H_;
    bcol[j] = bias[n];
  }

  v8f acc[8];
#pragma unroll
  for (int j = 0; j < 8; ++j) acc[j] = v8f_zero();

  for (int k0 = 0; k0 < H_; k0 += 32) {
    v16bf a = frag_f32(arow, k0, hf);   // A cvt amortized over 8 WMMAs
#pragma unroll
    for (int j = 0; j < 8; ++j) {
      v16bf b = frag_bf16(wrow[j], k0, hf);  // pure b128 loads, no cvt
      acc[j] = wmma_bf16(a, b, acc[j]);
    }
  }

  const int bb = m0 / S_;
  const int sbase = (m0 % S_) + 8 * hf;  // 8 consecutive rows per lane in C frag
#pragma unroll
  for (int j = 0; j < 8; ++j) {
    const int n = n0 + j * 16 + l16;
    const int hh = n / HD_, d = n % HD_;
    if (z == 2) {
      // V transposed [b,h,d,s]: 8 consecutive s -> one 16B store
      v8bf pack;
#pragma unroll
      for (int r = 0; r < 8; ++r) pack[r] = (__bf16)(acc[j][r] + bcol[j]);
      *(v8bf*)(vtw + ((((size_t)bb * NH_ + hh) * HD_ + d) * S_ + sbase)) = pack;
    } else {
      __bf16* dst = (z == 0) ? qw : kw;
      const size_t rb = ((size_t)bb * NH_ + hh) * S_;
#pragma unroll
      for (int r = 0; r < 8; ++r)
        dst[(rb + sbase + r) * HD_ + d] = (__bf16)(acc[j][r] + bcol[j]);
    }
  }
}

// ---------------------------------------------------------------------------
// Kernel 2: flash attention per (b, h, 128-query tile); each wave owns 16 rows.
// Streams 32-key tiles: 4 WMMAs scores (K=32 x 2 over HD=64, x 2 key subtiles),
// online softmax (shfl_xor row reductions in 16-lane halves), P via per-wave
// LDS transpose (C-layout -> A-layout), 4 WMMAs for P*V.
// O written bf16 to [b, s, h*HD+d]  (= [B,S,H] merged-head layout).
// ---------------------------------------------------------------------------
__global__ __launch_bounds__(256) void flash_attn_kernel(
    const __bf16* __restrict__ qw, const __bf16* __restrict__ kw,
    const __bf16* __restrict__ vtw, const unsigned char* __restrict__ mask,
    const int* __restrict__ seq_mask, __bf16* __restrict__ ow) {
  const int tid = threadIdx.x, wave = tid >> 5, lane = tid & 31;
  const int hf = lane >> 4, l16 = lane & 15;
  const int b = blockIdx.z, h = blockIdx.y;
  const int q0 = blockIdx.x * 128 + wave * 16;
  const int causal = seq_mask[0];
  const float sc = 0.03125f;  // 1/sqrt(H) = 1/32

  const __bf16* Qb = qw + ((size_t)b * NH_ + h) * S_ * HD_;
  const __bf16* Kb = kw + ((size_t)b * NH_ + h) * S_ * HD_;
  const __bf16* Vt = vtw + ((size_t)b * NH_ + h) * HD_ * S_;
  const unsigned char* mrow = mask + (size_t)b * S_;

  __shared__ __align__(16) __bf16 Pbuf[8][16][40];  // per-wave P transpose buffer

  const __bf16* qrowp = Qb + (size_t)(q0 + l16) * HD_;
  const v16bf qa0 = frag_bf16(qrowp, 0, hf);
  const v16bf qa1 = frag_bf16(qrowp, 32, hf);

  v8f o[4];
#pragma unroll
  for (int j = 0; j < 4; ++j) o[j] = v8f_zero();
  float mst[8], lst[8];
#pragma unroll
  for (int r = 0; r < 8; ++r) { mst[r] = -3.0e38f; lst[r] = 0.0f; }

  for (int kb = 0; kb < S_; kb += 32) {
    // ---- scores S = Q K^T  (16 rows x 32 keys, fp32) ----
    const __bf16* kr0 = Kb + (size_t)(kb + l16) * HD_;
    const __bf16* kr1 = Kb + (size_t)(kb + 16 + l16) * HD_;
    v8f c0 = v8f_zero(), c1 = v8f_zero();
    c0 = wmma_bf16(qa0, frag_bf16(kr0, 0, hf), c0);
    c0 = wmma_bf16(qa1, frag_bf16(kr0, 32, hf), c0);
    c1 = wmma_bf16(qa0, frag_bf16(kr1, 0, hf), c1);
    c1 = wmma_bf16(qa1, frag_bf16(kr1, 32, hf), c1);

    const int key0 = kb + l16, key1 = kb + 16 + l16;
    const bool mk0 = mrow[key0] != 0;
    const bool mk1 = mrow[key1] != 0;

    float resc[8];
#pragma unroll
    for (int r = 0; r < 8; ++r) {
      const int qrow = q0 + 8 * hf + r;
      float s0 = c0[r] * sc + ((mk0 || (causal && key0 > qrow)) ? NEGV : 0.0f);
      float s1 = c1[r] * sc + ((mk1 || (causal && key1 > qrow)) ? NEGV : 0.0f);
      // row max across the 16 columns (lanes of this half)
      float t = fmaxf(s0, s1);
#pragma unroll
      for (int xm = 1; xm < 16; xm <<= 1) t = fmaxf(t, __shfl_xor(t, xm, 32));
      const float mn = fmaxf(mst[r], t);
      resc[r] = __expf(mst[r] - mn);
      const float p0 = __expf(s0 - mn);
      const float p1 = __expf(s1 - mn);
      float rs = p0 + p1;
#pragma unroll
      for (int xm = 1; xm < 16; xm <<= 1) rs += __shfl_xor(rs, xm, 32);
      lst[r] = lst[r] * resc[r] + rs;
      mst[r] = mn;
      c0[r] = p0;
      c1[r] = p1;
    }

    // rescale running O
#pragma unroll
    for (int j = 0; j < 4; ++j)
#pragma unroll
      for (int r = 0; r < 8; ++r) o[j][r] *= resc[r];

    // P: C-fragment layout -> LDS -> A-fragment layout (per-wave private)
#pragma unroll
    for (int r = 0; r < 8; ++r) {
      Pbuf[wave][8 * hf + r][l16]      = (__bf16)c0[r];
      Pbuf[wave][8 * hf + r][16 + l16] = (__bf16)c1[r];
    }
    asm volatile("s_wait_dscnt 0x0" ::: "memory");

    v16bf pa;
#pragma unroll
    for (int i = 0; i < 8; ++i) {
      pa[i]     = Pbuf[wave][l16][hf * 8 + i];
      pa[8 + i] = Pbuf[wave][l16][16 + hf * 8 + i];
    }

    // ---- O += P * V  (V transposed layout -> contiguous B-fragment loads) ----
#pragma unroll
    for (int j = 0; j < 4; ++j) {
      v16bf vb = frag_bf16(Vt + (size_t)(j * 16 + l16) * S_, kb, hf);
      o[j] = wmma_bf16(pa, vb, o[j]);
    }
    asm volatile("s_wait_dscnt 0x0" ::: "memory");  // drain reads before overwrite
  }

  // normalize and store bf16 to merged-head [B,S,H]
#pragma unroll
  for (int r = 0; r < 8; ++r) {
    const float inv = 1.0f / lst[r];
#pragma unroll
    for (int j = 0; j < 4; ++j) o[j][r] *= inv;
  }
#pragma unroll
  for (int j = 0; j < 4; ++j) {
    const int col = h * HD_ + j * 16 + l16;
#pragma unroll
    for (int r = 0; r < 8; ++r) {
      const int s = q0 + 8 * hf + r;
      ow[((size_t)b * S_ + s) * H_ + col] = (__bf16)o[j][r];
    }
  }
}

// ---------------------------------------------------------------------------
// Kernel 3: output projection  out = O @ Wo^T + bo   (fp32 result)
// A (attention output) and Wo are both bf16 -> zero cvt in inner loop.
// ---------------------------------------------------------------------------
__global__ __launch_bounds__(256) void out_proj_kernel(
    const __bf16* __restrict__ ow, const __bf16* __restrict__ Wo,
    const float* __restrict__ bo, float* __restrict__ out) {
  const int tid = threadIdx.x, wave = tid >> 5, lane = tid & 31;
  const int hf = lane >> 4, l16 = lane & 15;
  const int m0 = blockIdx.x * 128 + wave * 16;
  const int n0 = blockIdx.y * 128;

  const __bf16* arow = ow + (size_t)(m0 + l16) * H_;

  const __bf16* wrow[8];
  float bcol[8];
#pragma unroll
  for (int j = 0; j < 8; ++j) {
    int n = n0 + j * 16 + l16;
    wrow[j] = Wo + (size_t)n * H_;
    bcol[j] = bo[n];
  }

  v8f acc[8];
#pragma unroll
  for (int j = 0; j < 8; ++j) acc[j] = v8f_zero();

  for (int k0 = 0; k0 < H_; k0 += 32) {
    v16bf a = frag_bf16(arow, k0, hf);
#pragma unroll
    for (int j = 0; j < 8; ++j) {
      v16bf b = frag_bf16(wrow[j], k0, hf);
      acc[j] = wmma_bf16(a, b, acc[j]);
    }
  }

  const int rbase = m0 + 8 * hf;
#pragma unroll
  for (int j = 0; j < 8; ++j) {
    const int n = n0 + j * 16 + l16;
#pragma unroll
    for (int r = 0; r < 8; ++r)
      out[(size_t)(rbase + r) * H_ + n] = acc[j][r] + bcol[j];
  }
}

// ---------------------------------------------------------------------------
// Host launcher
// ---------------------------------------------------------------------------
extern "C" void kernel_launch(void* const* d_in, const int* in_sizes, int n_in,
                              void* d_out, int out_size, void* d_ws, size_t ws_size,
                              hipStream_t stream) {
  const float* q_in = (const float*)d_in[0];
  const float* k_in = (const float*)d_in[1];
  const float* v_in = (const float*)d_in[2];
  const unsigned char* mask = (const unsigned char*)d_in[3];
  const int* seqm = (const int*)d_in[4];
  const float* Wq = (const float*)d_in[5];
  const float* bq = (const float*)d_in[6];
  const float* Wk = (const float*)d_in[7];
  const float* bk = (const float*)d_in[8];
  const float* Wv = (const float*)d_in[9];
  const float* bv = (const float*)d_in[10];
  const float* Wo = (const float*)d_in[11];
  const float* bo = (const float*)d_in[12];
  float* out = (float*)d_out;

  // bf16 workspace:
  //   Q[b,h,s,d] | K[b,h,s,d] | V^T[b,h,d,s] | O[b,s,H]   (16MB each)
  //   Wq | Wk | Wv | Wo  bf16                             (2MB each)
  const size_t act_bytes = (size_t)B_ * NH_ * S_ * HD_ * sizeof(uint16_t);  // 16MB
  const size_t w_elems = (size_t)H_ * H_;                                   // 1M
  const size_t w_bytes = w_elems * sizeof(uint16_t);                        // 2MB
  char* wsb = (char*)d_ws;
  __bf16* qw   = (__bf16*)(wsb);
  __bf16* kw   = (__bf16*)(wsb + act_bytes);
  __bf16* vtw  = (__bf16*)(wsb + 2 * act_bytes);
  __bf16* oww  = (__bf16*)(wsb + 3 * act_bytes);
  __bf16* wq_b = (__bf16*)(wsb + 4 * act_bytes);
  __bf16* wk_b = (__bf16*)(wsb + 4 * act_bytes + w_bytes);
  __bf16* wv_b = (__bf16*)(wsb + 4 * act_bytes + 2 * w_bytes);
  __bf16* wo_b = (__bf16*)(wsb + 4 * act_bytes + 3 * w_bytes);

  const int cvt_blocks = (int)(w_elems / (256 * 8));  // 512
  cvt_f32_to_bf16_kernel<<<cvt_blocks, 256, 0, stream>>>(Wq, wq_b);
  cvt_f32_to_bf16_kernel<<<cvt_blocks, 256, 0, stream>>>(Wk, wk_b);
  cvt_f32_to_bf16_kernel<<<cvt_blocks, 256, 0, stream>>>(Wv, wv_b);
  cvt_f32_to_bf16_kernel<<<cvt_blocks, 256, 0, stream>>>(Wo, wo_b);

  qkv_proj_kernel<<<dim3((B_ * S_) / 128, H_ / 128, 3), 256, 0, stream>>>(
      q_in, k_in, v_in, wq_b, bq, wk_b, bk, wv_b, bv, qw, kw, vtw);
  flash_attn_kernel<<<dim3(S_ / 128, NH_, B_), 256, 0, stream>>>(
      qw, kw, vtw, mask, seqm, oww);
  out_proj_kernel<<<dim3((B_ * S_) / 128, H_ / 128), 256, 0, stream>>>(
      oww, wo_b, bo, out);
}
